// MambaBlock_1743756722395
// MI455X (gfx1250) — matrix operand order repaired and checked by reading, not compiled, not run on hardware
//
#include <hip/hip_runtime.h>

#define D_MODEL 1024
#define D_STATE 16
#define D_CONV  4
#define D_INNER 2048
#define DT_RANK 64
#define BATCH   4
#define SEQ     2048
#define NROWS   (BATCH * SEQ)           /* 8192 token rows */
#define N_XPROJ (DT_RANK + 2 * D_STATE) /* 96 */
#define DBL_LD  128                     /* padded leading dim of x_proj output */
#define NCHUNK  16
#define CHUNK   (SEQ / NCHUNK)          /* 128 */

typedef __attribute__((ext_vector_type(16))) __bf16 v16bf;
typedef __attribute__((ext_vector_type(8)))  float  v8f;

// ---------------------------------------------------------------------------
// helpers
// ---------------------------------------------------------------------------
__device__ __forceinline__ unsigned short f2bf(float f) {
  union { float f; unsigned int u; } v; v.f = f;
  unsigned int u = v.u;
  unsigned int r = u + 0x7FFFu + ((u >> 16) & 1u);  // RNE
  return (unsigned short)(r >> 16);
}

__device__ __forceinline__ float silu_f(float x) {
  return x / (1.0f + __expf(-x));
}

// scan chunk-state layout: [(chunk, state), batch, channel] so adjacent
// channels (adjacent lanes) are contiguous in every pass
__device__ __forceinline__ size_t pq_idx(int b, int d, int c, int n) {
  return (((size_t)c * D_STATE + n) * BATCH + b) * D_INNER + d;
}

// ---------------------------------------------------------------------------
// LayerNorm over D_MODEL, emits bf16 activations for the WMMA GEMM
// ---------------------------------------------------------------------------
__global__ void __launch_bounds__(256)
ln_kernel(const float* __restrict__ x, const float* __restrict__ w,
          const float* __restrict__ b, unsigned short* __restrict__ xn_bf) {
  const int row = blockIdx.x;
  const size_t base = (size_t)row * D_MODEL;
  __shared__ float s1[256];
  __shared__ float s2[256];
  const int tid = threadIdx.x;
  float a = 0.f, q = 0.f;
  for (int i = tid; i < D_MODEL; i += 256) {
    float v = x[base + i];
    a += v; q += v * v;
  }
  s1[tid] = a; s2[tid] = q;
  __syncthreads();
  for (int off = 128; off > 0; off >>= 1) {
    if (tid < off) { s1[tid] += s1[tid + off]; s2[tid] += s2[tid + off]; }
    __syncthreads();
  }
  const float mu   = s1[0] * (1.0f / D_MODEL);
  const float var  = s2[0] * (1.0f / D_MODEL) - mu * mu;
  const float rstd = rsqrtf(var + 1e-5f);
  for (int i = tid; i < D_MODEL; i += 256) {
    float v = (x[base + i] - mu) * rstd * w[i] + b[i];
    xn_bf[base + i] = f2bf(v);
  }
}

// ---------------------------------------------------------------------------
// f32 -> bf16 conversion with zero padding (weights)
// ---------------------------------------------------------------------------
__global__ void __launch_bounds__(256)
cvt_bf16_kernel(const float* __restrict__ in, unsigned short* __restrict__ out,
                int n, int npad) {
  int i = blockIdx.x * 256 + threadIdx.x;
  if (i < npad) out[i] = (i < n) ? f2bf(in[i]) : (unsigned short)0;
}

// ---------------------------------------------------------------------------
// WMMA GEMM: C[M,N] = A[M,K](bf16) * W[N,K](bf16)^T (+ resid), f32 accumulate.
// block = 256 thr (8 waves). Block tile 256(M) x 64(N), K step 32.
// Each wave computes 32x64 -> 8 v_wmma per K step. Double-buffered LDS,
// one barrier per K step. No guards: M % 256 == 0, K % 32 == 0,
// W padded to gridDim.x*64 rows, C padded to ldc columns.
// ---------------------------------------------------------------------------
__global__ void __launch_bounds__(256)
gemm_bf16_wmma(const unsigned short* __restrict__ A,
               const unsigned short* __restrict__ W,
               const float* __restrict__ resid,
               float* __restrict__ C,
               int K, int ldc) {
  // row stride 20 dwords (80 B): 16B-aligned quads, 20-bank rotation
  __shared__ unsigned int sA[2][256 * 20];   // 2 x 20 KB
  __shared__ unsigned int sB[2][64 * 20];    // 2 x 5 KB
  const int tid  = threadIdx.x;
  const int lane = tid & 31;
  const int wave = tid >> 5;
  const int half = lane >> 4;
  const int l16  = lane & 15;
  const int rowBase = blockIdx.y * 256;
  const int colBase = blockIdx.x * 64;
  const uint4* Aq = reinterpret_cast<const uint4*>(A);
  const uint4* Wq = reinterpret_cast<const uint4*>(W);
  const int ldq = K >> 3;                 // row stride in uint4 units

  // staging coordinates: one b128 per row-quarter
  const int arow = tid >> 2;              // 0..63
  const int quad = tid & 3;
  const size_t aoff0 = (size_t)(rowBase + arow)       * ldq + quad;
  const size_t aoff1 = (size_t)(rowBase + arow + 64)  * ldq + quad;
  const size_t aoff2 = (size_t)(rowBase + arow + 128) * ldq + quad;
  const size_t aoff3 = (size_t)(rowBase + arow + 192) * ldq + quad;
  const size_t boff  = (size_t)(colBase + arow)       * ldq + quad;
  const int sA0 = (arow)       * 20 + quad * 4;
  const int sA1 = (arow + 64)  * 20 + quad * 4;
  const int sA2 = (arow + 128) * 20 + quad * 4;
  const int sA3 = (arow + 192) * 20 + quad * 4;
  const int sB0 = (arow)       * 20 + quad * 4;

  v8f czero = {};
  v8f acc[2][4];
#pragma unroll
  for (int mi = 0; mi < 2; ++mi)
#pragma unroll
    for (int ns = 0; ns < 4; ++ns) acc[mi][ns] = czero;

  const int mRow0 = wave * 32 + l16;
  const int mRow1 = mRow0 + 16;
  const int ktiles = K >> 5;

  // prologue: stage tile 0
  uint4 ra0 = Aq[aoff0], ra1 = Aq[aoff1], ra2 = Aq[aoff2], ra3 = Aq[aoff3];
  uint4 rb0 = Wq[boff];
  *reinterpret_cast<uint4*>(&sA[0][sA0]) = ra0;
  *reinterpret_cast<uint4*>(&sA[0][sA1]) = ra1;
  *reinterpret_cast<uint4*>(&sA[0][sA2]) = ra2;
  *reinterpret_cast<uint4*>(&sA[0][sA3]) = ra3;
  *reinterpret_cast<uint4*>(&sB[0][sB0]) = rb0;
  __syncthreads();

  for (int kt = 0; kt < ktiles; ++kt) {
    const int cur = kt & 1;
    const bool more = (kt + 1) < ktiles;
    if (more) {  // issue next tile's global loads; they overlap the WMMAs
      const int kq = (kt + 1) * 4;
      ra0 = Aq[aoff0 + kq]; ra1 = Aq[aoff1 + kq];
      ra2 = Aq[aoff2 + kq]; ra3 = Aq[aoff3 + kq];
      rb0 = Wq[boff + kq];
      if (kt + 2 < ktiles)
        __builtin_prefetch(&Aq[aoff0 + kq + 4], 0, 3);
    }

    const unsigned int* cA = sA[cur];
    const unsigned int* cB = sB[cur];
    // A fragments: VGPR r holds K dword (r<4?0:8) + half*4 + (r&3)
    union { v16bf v; unsigned int u[8]; } af0, af1;
#pragma unroll
    for (int r = 0; r < 8; ++r) {
      int kd = ((r & 4) ? 8 : 0) + half * 4 + (r & 3);
      af0.u[r] = cA[mRow0 * 20 + kd];
      af1.u[r] = cA[mRow1 * 20 + kd];
    }
    // B fragments: lane = column, VGPR r holds K dword (r<4?0:8) + half*4 + (r&3)
    union { v16bf v; unsigned int u[8]; } bf0, bf1, bf2, bf3;
#pragma unroll
    for (int r = 0; r < 8; ++r) {
      int kd = ((r & 4) ? 8 : 0) + half * 4 + (r & 3);
      bf0.u[r] = cB[(0 * 16 + l16) * 20 + kd];
      bf1.u[r] = cB[(1 * 16 + l16) * 20 + kd];
      bf2.u[r] = cB[(2 * 16 + l16) * 20 + kd];
      bf3.u[r] = cB[(3 * 16 + l16) * 20 + kd];
    }
    acc[0][0] = __builtin_amdgcn_wmma_f32_16x16x32_bf16(false, af0.v, false, bf0.v, (short)0, acc[0][0], false, false);
    acc[0][1] = __builtin_amdgcn_wmma_f32_16x16x32_bf16(false, af0.v, false, bf1.v, (short)0, acc[0][1], false, false);
    acc[0][2] = __builtin_amdgcn_wmma_f32_16x16x32_bf16(false, af0.v, false, bf2.v, (short)0, acc[0][2], false, false);
    acc[0][3] = __builtin_amdgcn_wmma_f32_16x16x32_bf16(false, af0.v, false, bf3.v, (short)0, acc[0][3], false, false);
    acc[1][0] = __builtin_amdgcn_wmma_f32_16x16x32_bf16(false, af1.v, false, bf0.v, (short)0, acc[1][0], false, false);
    acc[1][1] = __builtin_amdgcn_wmma_f32_16x16x32_bf16(false, af1.v, false, bf1.v, (short)0, acc[1][1], false, false);
    acc[1][2] = __builtin_amdgcn_wmma_f32_16x16x32_bf16(false, af1.v, false, bf2.v, (short)0, acc[1][2], false, false);
    acc[1][3] = __builtin_amdgcn_wmma_f32_16x16x32_bf16(false, af1.v, false, bf3.v, (short)0, acc[1][3], false, false);

    if (more) {  // publish next tile into the other buffer; single barrier
      unsigned int* dA = sA[cur ^ 1];
      unsigned int* dB = sB[cur ^ 1];
      *reinterpret_cast<uint4*>(&dA[sA0]) = ra0;
      *reinterpret_cast<uint4*>(&dA[sA1]) = ra1;
      *reinterpret_cast<uint4*>(&dA[sA2]) = ra2;
      *reinterpret_cast<uint4*>(&dA[sA3]) = ra3;
      *reinterpret_cast<uint4*>(&dB[sB0]) = rb0;
      __syncthreads();
    }
  }

  // epilogue: C/D layout — VGPR v: M = v + (lane>>4)*8, N = lane&15
  // uniform branch on resid; stores unguarded (buffers tile-padded)
  if (resid != nullptr) {
#pragma unroll
    for (int mi = 0; mi < 2; ++mi) {
      const int mBase = rowBase + wave * 32 + mi * 16 + half * 8;
#pragma unroll
      for (int ns = 0; ns < 4; ++ns) {
        const int gCol = colBase + ns * 16 + l16;
#pragma unroll
        for (int v = 0; v < 8; ++v) {
          size_t idx = (size_t)(mBase + v) * ldc + gCol;
          C[idx] = acc[mi][ns][v] + resid[idx];
        }
      }
    }
  } else {
#pragma unroll
    for (int mi = 0; mi < 2; ++mi) {
      const int mBase = rowBase + wave * 32 + mi * 16 + half * 8;
#pragma unroll
      for (int ns = 0; ns < 4; ++ns) {
        const int gCol = colBase + ns * 16 + l16;
#pragma unroll
        for (int v = 0; v < 8; ++v) {
          size_t idx = (size_t)(mBase + v) * ldc + gCol;
          C[idx] = acc[mi][ns][v];
        }
      }
    }
  }
}

// ---------------------------------------------------------------------------
// causal depthwise conv1d (D_CONV=4) + silu; reads u-half of xz
// ---------------------------------------------------------------------------
__global__ void __launch_bounds__(256)
conv_silu_kernel(const float* __restrict__ xz, const float* __restrict__ conv_w,
                 const float* __restrict__ conv_b,
                 float* __restrict__ u, unsigned short* __restrict__ u_bf) {
  size_t idx = (size_t)blockIdx.x * 256 + threadIdx.x;
  if (idx >= (size_t)NROWS * D_INNER) return;
  const int d   = (int)(idx % D_INNER);
  const long bl = (long)(idx / D_INNER);   // = b*SEQ + l
  const int l   = (int)(bl % SEQ);
  float acc = conv_b[d];
#pragma unroll
  for (int j = 0; j < D_CONV; ++j) {
    int ls = l - (D_CONV - 1) + j;
    if (ls >= 0)
      acc += conv_w[d * D_CONV + j] *
             xz[(size_t)(bl + (ls - l)) * (2 * D_INNER) + d];
  }
  float s = silu_f(acc);
  u[idx]    = s;
  u_bf[idx] = f2bf(s);
}

// ---------------------------------------------------------------------------
// extract dt columns (0..63) of dbl (ld = DBL_LD) as bf16
// ---------------------------------------------------------------------------
__global__ void __launch_bounds__(256)
dt_extract_kernel(const float* __restrict__ dbl, unsigned short* __restrict__ dt_bf) {
  size_t idx = (size_t)blockIdx.x * 256 + threadIdx.x;
  if (idx >= (size_t)NROWS * DT_RANK) return;
  size_t m = idx / DT_RANK;
  int    r = (int)(idx % DT_RANK);
  dt_bf[idx] = f2bf(dbl[m * DBL_LD + r]);
}

// ---------------------------------------------------------------------------
// delta = softplus(delta_pre + dt_proj_b[d]), in place
// ---------------------------------------------------------------------------
__global__ void __launch_bounds__(256)
softplus_bias_kernel(float* __restrict__ delta, const float* __restrict__ bias) {
  size_t idx = (size_t)blockIdx.x * 256 + threadIdx.x;
  if (idx >= (size_t)NROWS * D_INNER) return;
  const int d = (int)(idx % D_INNER);
  float v = delta[idx] + bias[d];
  delta[idx] = (v > 20.0f) ? v : __logf(1.0f + __expf(v));
}

// ---------------------------------------------------------------------------
// Chunked selective scan (h_l = dA_l * h_{l-1} + dB_l is associative).
// Pass 1: per 128-step chunk compute transfer (P = prod dA, Q = state from 0).
// Pass 2: serial combine over the 16 chunks -> initial state per chunk.
// Pass 3: replay each chunk from its true initial state; fuse D*u skip and
//         silu(z) gate; emit bf16 for the out_proj WMMA.
// ---------------------------------------------------------------------------
__global__ void __launch_bounds__(256)
scan_pass1(const float* __restrict__ delta, const float* __restrict__ u,
           const float* __restrict__ dbl, const float* __restrict__ A_log,
           float* __restrict__ P, float* __restrict__ Q) {
  const int b = blockIdx.y, c = blockIdx.z;
  const int d = blockIdx.x * 256 + threadIdx.x;
  __shared__ float sB[CHUNK * D_STATE];          // 8 KB

#pragma unroll
  for (int i = 0; i < 8; ++i) {
    int e = threadIdx.x + i * 256;
    int l = e >> 4, n = e & 15;
    sB[e] = dbl[((size_t)b * SEQ + c * CHUNK + l) * DBL_LD + DT_RANK + n];
  }
  __syncthreads();

  float Avec[D_STATE], h[D_STATE], p[D_STATE];
#pragma unroll
  for (int n = 0; n < D_STATE; ++n) {
    Avec[n] = -__expf(A_log[d * D_STATE + n]);
    h[n] = 0.f; p[n] = 1.f;
  }
  for (int l = 0; l < CHUNK; ++l) {
    const size_t row = (size_t)b * SEQ + c * CHUNK + l;
    const float dt = delta[row * D_INNER + d];
    const float du = dt * u[row * D_INNER + d];
    const float* bl = &sB[l * D_STATE];
#pragma unroll
    for (int n = 0; n < D_STATE; ++n) {
      float dA = __expf(dt * Avec[n]);
      h[n] = dA * h[n] + du * bl[n];
      p[n] *= dA;
    }
  }
#pragma unroll
  for (int n = 0; n < D_STATE; ++n) {
    P[pq_idx(b, d, c, n)] = p[n];
    Q[pq_idx(b, d, c, n)] = h[n];
  }
}

__global__ void __launch_bounds__(256)
scan_pass2(const float* __restrict__ P, const float* __restrict__ Q,
           float* __restrict__ Hc) {
  const int b = blockIdx.y;
  const int d = blockIdx.x * 256 + threadIdx.x;
  float H[D_STATE];
#pragma unroll
  for (int n = 0; n < D_STATE; ++n) H[n] = 0.f;
  for (int c = 0; c < NCHUNK; ++c) {
#pragma unroll
    for (int n = 0; n < D_STATE; ++n) {
      const size_t ix = pq_idx(b, d, c, n);
      Hc[ix] = H[n];
      H[n] = P[ix] * H[n] + Q[ix];
    }
  }
}

__global__ void __launch_bounds__(256)
scan_pass3(const float* __restrict__ delta, const float* __restrict__ u,
           const float* __restrict__ dbl, const float* __restrict__ xz,
           const float* __restrict__ A_log, const float* __restrict__ Dp,
           const float* __restrict__ Hc, unsigned short* __restrict__ y_bf) {
  const int b = blockIdx.y, c = blockIdx.z;
  const int d = blockIdx.x * 256 + threadIdx.x;
  __shared__ float sB[CHUNK * D_STATE];          // 8 KB
  __shared__ float sC[CHUNK * D_STATE];          // 8 KB

#pragma unroll
  for (int i = 0; i < 8; ++i) {
    int e = threadIdx.x + i * 256;
    int l = e >> 4, n = e & 15;
    const size_t row = (size_t)b * SEQ + c * CHUNK + l;
    sB[e] = dbl[row * DBL_LD + DT_RANK + n];
    sC[e] = dbl[row * DBL_LD + DT_RANK + D_STATE + n];
  }
  __syncthreads();

  float Avec[D_STATE], h[D_STATE];
#pragma unroll
  for (int n = 0; n < D_STATE; ++n) {
    Avec[n] = -__expf(A_log[d * D_STATE + n]);
    h[n] = Hc[pq_idx(b, d, c, n)];
  }
  const float Dd = Dp[d];

  for (int l = 0; l < CHUNK; ++l) {
    const size_t row = (size_t)b * SEQ + c * CHUNK + l;
    const float dt = delta[row * D_INNER + d];
    const float uu = u[row * D_INNER + d];
    const float du = dt * uu;
    const float* bl = &sB[l * D_STATE];
    const float* cl = &sC[l * D_STATE];
    float y = 0.f;
#pragma unroll
    for (int n = 0; n < D_STATE; ++n) {
      float dA = __expf(dt * Avec[n]);
      h[n] = dA * h[n] + du * bl[n];
      y += h[n] * cl[n];
    }
    const float zz  = xz[row * (2 * D_INNER) + D_INNER + d];
    const float out = (y + uu * Dd) * silu_f(zz);
    y_bf[row * D_INNER + d] = f2bf(out);
  }
}

// ---------------------------------------------------------------------------
// driver
// ---------------------------------------------------------------------------
extern "C" void kernel_launch(void* const* d_in, const int* in_sizes, int n_in,
                              void* d_out, int out_size, void* d_ws, size_t ws_size,
                              hipStream_t stream) {
  const float* x          = (const float*)d_in[0];
  const float* ln_w       = (const float*)d_in[1];
  const float* ln_b       = (const float*)d_in[2];
  const float* in_proj_w  = (const float*)d_in[3];
  const float* conv_w     = (const float*)d_in[4];
  const float* conv_b     = (const float*)d_in[5];
  const float* x_proj_w   = (const float*)d_in[6];
  const float* dt_proj_w  = (const float*)d_in[7];
  const float* dt_proj_b  = (const float*)d_in[8];
  const float* A_log      = (const float*)d_in[9];
  const float* Dp         = (const float*)d_in[10];
  const float* out_proj_w = (const float*)d_in[11];
  float* out = (float*)d_out;

  char* ws = (char*)d_ws;
  size_t off = 0;
  auto carve = [&](size_t bytes) -> void* {
    void* p = ws + off;
    off += (bytes + 255) & ~(size_t)255;
    return p;
  };

  unsigned short* xn_bf = (unsigned short*)carve((size_t)NROWS * D_MODEL * 2);
  unsigned short* w1_bf = (unsigned short*)carve((size_t)2 * D_INNER * D_MODEL * 2);
  unsigned short* w2_bf = (unsigned short*)carve((size_t)DBL_LD * D_INNER * 2);
  unsigned short* w3_bf = (unsigned short*)carve((size_t)D_INNER * DT_RANK * 2);
  unsigned short* w4_bf = (unsigned short*)carve((size_t)D_MODEL * D_INNER * 2);
  float*          xz    = (float*)carve((size_t)NROWS * 2 * D_INNER * 4);
  float*          u     = (float*)carve((size_t)NROWS * D_INNER * 4);
  unsigned short* u_bf  = (unsigned short*)carve((size_t)NROWS * D_INNER * 2);
  float*          dbl   = (float*)carve((size_t)NROWS * DBL_LD * 4);
  unsigned short* dt_bf = (unsigned short*)carve((size_t)NROWS * DT_RANK * 2);
  float*          delta = (float*)carve((size_t)NROWS * D_INNER * 4);
  unsigned short* y_bf  = (unsigned short*)carve((size_t)NROWS * D_INNER * 2);
  const size_t nscan = (size_t)BATCH * D_INNER * NCHUNK * D_STATE;
  float*          Pbuf  = (float*)carve(nscan * 4);
  float*          Qbuf  = (float*)carve(nscan * 4);
  float*          Hbuf  = (float*)carve(nscan * 4);

  // 1. LayerNorm -> bf16
  ln_kernel<<<NROWS, 256, 0, stream>>>(x, ln_w, ln_b, xn_bf);

  // 2. weights -> bf16 (x_proj padded to 128 rows so GEMM is guard-free)
  {
    int n1 = 2 * D_INNER * D_MODEL;
    int n2 = N_XPROJ * D_INNER, n2p = DBL_LD * D_INNER;
    int n3 = D_INNER * DT_RANK, n4 = D_MODEL * D_INNER;
    cvt_bf16_kernel<<<(n1 + 255) / 256, 256, 0, stream>>>(in_proj_w,  w1_bf, n1, n1);
    cvt_bf16_kernel<<<(n2p + 255) / 256, 256, 0, stream>>>(x_proj_w,  w2_bf, n2, n2p);
    cvt_bf16_kernel<<<(n3 + 255) / 256, 256, 0, stream>>>(dt_proj_w,  w3_bf, n3, n3);
    cvt_bf16_kernel<<<(n4 + 255) / 256, 256, 0, stream>>>(out_proj_w, w4_bf, n4, n4);
  }

  // 3. in_proj: xz[8192,4096] = xn * W1^T
  {
    dim3 g((2 * D_INNER) / 64, NROWS / 256);
    gemm_bf16_wmma<<<g, 256, 0, stream>>>(xn_bf, w1_bf, nullptr, xz,
                                          D_MODEL, 2 * D_INNER);
  }

  // 4. depthwise conv + silu
  {
    size_t n = (size_t)NROWS * D_INNER;
    conv_silu_kernel<<<(unsigned)((n + 255) / 256), 256, 0, stream>>>(
        xz, conv_w, conv_b, u, u_bf);
  }

  // 5. x_proj: dbl[8192,128(pad)] = u * W2^T
  {
    dim3 g(DBL_LD / 64, NROWS / 256);
    gemm_bf16_wmma<<<g, 256, 0, stream>>>(u_bf, w2_bf, nullptr, dbl,
                                          D_INNER, DBL_LD);
  }

  // 6. dt columns -> bf16
  {
    size_t n = (size_t)NROWS * DT_RANK;
    dt_extract_kernel<<<(unsigned)((n + 255) / 256), 256, 0, stream>>>(dbl, dt_bf);
  }

  // 7. dt_proj: delta_pre[8192,2048] = dt * W3^T
  {
    dim3 g(D_INNER / 64, NROWS / 256);
    gemm_bf16_wmma<<<g, 256, 0, stream>>>(dt_bf, w3_bf, nullptr, delta,
                                          DT_RANK, D_INNER);
  }

  // 8. softplus(+bias)
  {
    size_t n = (size_t)NROWS * D_INNER;
    softplus_bias_kernel<<<(unsigned)((n + 255) / 256), 256, 0, stream>>>(delta, dt_proj_b);
  }

  // 9. chunked selective scan
  {
    dim3 g1(D_INNER / 256, BATCH, NCHUNK);
    scan_pass1<<<g1, 256, 0, stream>>>(delta, u, dbl, A_log, Pbuf, Qbuf);
    dim3 g2(D_INNER / 256, BATCH);
    scan_pass2<<<g2, 256, 0, stream>>>(Pbuf, Qbuf, Hbuf);
    scan_pass3<<<g1, 256, 0, stream>>>(delta, u, dbl, xz, A_log, Dp, Hbuf, y_bf);
  }

  // 10. out_proj + residual: out[8192,1024] = y * W4^T + x
  {
    dim3 g(D_MODEL / 64, NROWS / 256);
    gemm_bf16_wmma<<<g, 256, 0, stream>>>(y_bf, w4_bf, x, out,
                                          D_INNER, D_MODEL);
  }
}
